// MSA_80109730005335
// MI455X (gfx1250) — matrix operand order repaired and checked by reading, not compiled, not run on hardware
//
#include <hip/hip_runtime.h>

typedef __bf16 bf16_t;
typedef __attribute__((ext_vector_type(16))) bf16_t v16bf;
typedef __attribute__((ext_vector_type(8)))  bf16_t v8bf;
typedef __attribute__((ext_vector_type(8)))  float  v8f;
typedef __attribute__((ext_vector_type(4)))  unsigned int u32x4;
typedef __attribute__((ext_vector_type(8)))  int i32x8;
typedef __attribute__((ext_vector_type(4)))  int i32x4;

static __device__ inline v8f v8f_zero() {
  v8f z;
#pragma unroll
  for (int i = 0; i < 8; ++i) z[i] = 0.0f;
  return z;
}

static __device__ inline v16bf combine16(v8bf lo, v8bf hi) {
  v16bf r;
#pragma unroll
  for (int i = 0; i < 8; ++i) { r[i] = lo[i]; r[i + 8] = hi[i]; }
  return r;
}

// A fragment: 16x32 bf16, row-major source with leading dim lda (elements).
// Lane L (L<16): M=L, K in {0..7, 16..23}; lane L+16: M=L, K in {8..15, 24..31}.
static __device__ inline v16bf load_a_frag(const bf16_t* a, int lda) {
  int lane = threadIdx.x & 31;
  const bf16_t* p = a + (size_t)(lane & 15) * lda + ((lane >> 4) << 3);
  return combine16(*(const v8bf*)p, *(const v8bf*)(p + 16));
}

// B fragment: 32x16 bf16 from a K-major source B[k][n], contiguous in n,
// leading dim ldb. Lane%16 = N; lane L: K in {0..7,16..23}, lane L+16: K in {8..15,24..31}.
static __device__ inline v16bf load_b_frag_rowmajor(const bf16_t* bt, int ldb) {
  int lane = threadIdx.x & 31;
  const bf16_t* p = bt + (lane & 15) + (size_t)((lane >> 4) << 3) * ldb;
  v16bf r;
#pragma unroll
  for (int j = 0; j < 16; ++j) r[j] = p[(size_t)(j + ((j >> 3) << 3)) * ldb];
  return r;
}

// B fragment by gather: B[k][n] = src[n*stride + k]  (K^T in attention).
static __device__ inline v16bf load_b_frag_gather(const bf16_t* src, int stride) {
  int lane = threadIdx.x & 31;
  const bf16_t* p = src + (size_t)(lane & 15) * stride + ((lane >> 4) << 3);
  v16bf r;
#pragma unroll
  for (int j = 0; j < 16; ++j) r[j] = p[j + ((j >> 3) << 3)];
  return r;
}

static __device__ inline v8f wmma_bf16(v16bf a, v16bf b, v8f c) {
  return __builtin_amdgcn_wmma_f32_16x16x32_bf16(
      false, a, false, b, (short)0, c, false, false);
}

// ---------------------------------------------------------------------------
// TDM: async 2D tile load (bf16 elements) global -> LDS.
// Descriptor packing per CDNA5 ISA §8 (Tensor DMA Descriptor).
// ---------------------------------------------------------------------------
static __device__ inline void tdm_load_2d(const bf16_t* gptr, bf16_t* lptr,
                                          unsigned tile_n, unsigned tile_rows,
                                          unsigned long long row_stride,
                                          unsigned tdim0, unsigned tdim1) {
  unsigned long long ga = (unsigned long long)(uintptr_t)gptr;  // byte addr
  unsigned la = (unsigned)(uintptr_t)lptr;                      // LDS byte offset
  u32x4 g0;
  g0[0] = 1u;                                   // count=1, user descriptor
  g0[1] = la;                                   // lds_addr [63:32]
  g0[2] = (unsigned)ga;                         // global_addr [95:64]
  g0[3] = (unsigned)((ga >> 32) & 0x1FFFFFFu)   // global_addr [120:96]
          | (2u << 30);                         // type=2 ("image") [127:126]
  i32x8 g1;
  g1[0] = (int)(1u << 16);                      // data_size=1 -> 2 bytes
  g1[1] = (int)((tdim0 & 0xFFFFu) << 16);       // tensor_dim0[15:0] @ [63:48]
  g1[2] = (int)(((tdim0 >> 16) & 0xFFFFu) | ((tdim1 & 0xFFFFu) << 16));
  g1[3] = (int)(((tdim1 >> 16) & 0xFFFFu) | (tile_n << 16));   // tile_dim0
  g1[4] = (int)(tile_rows & 0xFFFFu);           // tile_dim1 (tile_dim2 = 0)
  g1[5] = (int)(unsigned)row_stride;            // tensor_dim0_stride[31:0]
  g1[6] = (int)(unsigned)((row_stride >> 32) & 0xFFFFu);       // [47:32]
  g1[7] = 0;
  i32x4 z4;
  z4[0] = 0; z4[1] = 0; z4[2] = 0; z4[3] = 0;   // groups 2/3 unused (2D tile)
  i32x8 z8;
#pragma unroll
  for (int i = 0; i < 8; ++i) z8[i] = 0;
  __builtin_amdgcn_tensor_load_to_lds(g0, g1, z4, z4, z8, 0);
}

// ---------------------------------------------------------------------------
// Kernel 1: fp32 -> bf16 elementwise convert
// ---------------------------------------------------------------------------
__global__ void msa_cvt_bf16(const float* __restrict__ in,
                             bf16_t* __restrict__ out, int n) {
  int i = blockIdx.x * blockDim.x + threadIdx.x;
  if (i < n) out[i] = (bf16_t)in[i];
}

// ---------------------------------------------------------------------------
// Kernel 2: fp32 [R,C] -> bf16 transposed [C,R] (LDS 32x32 tile)
// ---------------------------------------------------------------------------
__global__ void msa_transpose_cvt(const float* __restrict__ in,
                                  bf16_t* __restrict__ out, int R, int C) {
  __shared__ float tile[32][33];
  int c0 = blockIdx.x * 32, r0 = blockIdx.y * 32;
  int tx = threadIdx.x, ty = threadIdx.y;
#pragma unroll
  for (int j = 0; j < 4; ++j)
    tile[ty + 8 * j][tx] = in[(size_t)(r0 + ty + 8 * j) * C + (c0 + tx)];
  __syncthreads();
#pragma unroll
  for (int j = 0; j < 4; ++j)
    out[(size_t)(c0 + ty + 8 * j) * R + (r0 + tx)] = (bf16_t)tile[tx][ty + 8 * j];
}

// ---------------------------------------------------------------------------
// TDM-staged GEMM: block = 128 threads (4 waves), block tile 64x64.
// TDM double-buffers the 64x32 A panel and 32x64 B panel in LDS;
// wave0 issues + waits TENSORcnt, workgroup barrier publishes to all waves.
// All fragments hoisted ahead of the WMMAs so ds_loads clause together and
// the 4 WMMAs issue back-to-back on the matrix pipe.
// ---------------------------------------------------------------------------
__global__ void msa_qkv_gemm(const bf16_t* __restrict__ X,
                             const bf16_t* __restrict__ Wt,
                             bf16_t* __restrict__ QKV) {
  const int K = 1024, N = 3072;
  int wave = threadIdx.x >> 5, lane = threadIdx.x & 31;
  int mblk = blockIdx.y * 64;
  int m0 = mblk + wave * 16;
  int n0 = blockIdx.x * 64;
  __shared__ __align__(16) bf16_t lA[2][64 * 32];
  __shared__ __align__(16) bf16_t lB[2][32 * 64];
  bool issuer = (threadIdx.x < 32);

  if (issuer) {
    tdm_load_2d(X + (size_t)mblk * K, &lA[0][0], 32, 64, K, K, 4096);
    tdm_load_2d(Wt + n0, &lB[0][0], 64, 32, N, N, K);
    __builtin_amdgcn_s_wait_tensorcnt(0);
  }
  __syncthreads();

  v8f acc[4];
#pragma unroll
  for (int t = 0; t < 4; ++t) acc[t] = v8f_zero();

  int cur = 0;
  for (int kk = 0; kk < K; kk += 32) {
    int nxt = cur ^ 1;
    if (issuer && (kk + 32 < K)) {
      tdm_load_2d(X + (size_t)mblk * K + (kk + 32), &lA[nxt][0], 32, 64, K, K, 4096);
      tdm_load_2d(Wt + (size_t)(kk + 32) * N + n0, &lB[nxt][0], 64, 32, N, N, K);
    }
    v16bf a = load_a_frag(&lA[cur][wave * 16 * 32], 32);
    v16bf b[4];
#pragma unroll
    for (int t = 0; t < 4; ++t) b[t] = load_b_frag_rowmajor(&lB[cur][16 * t], 64);
#pragma unroll
    for (int t = 0; t < 4; ++t) acc[t] = wmma_bf16(a, b[t], acc[t]);
    if (issuer) __builtin_amdgcn_s_wait_tensorcnt(0);
    __syncthreads();
    cur = nxt;
  }
  int hh = lane >> 4, nn = lane & 15;
#pragma unroll
  for (int t = 0; t < 4; ++t)
#pragma unroll
    for (int r = 0; r < 8; ++r)
      QKV[(size_t)(m0 + r + 8 * hh) * N + n0 + 16 * t + nn] = (bf16_t)acc[t][r];
}

__global__ void msa_out_gemm(const bf16_t* __restrict__ A,
                             const bf16_t* __restrict__ Wt,
                             const float* __restrict__ bias,
                             float* __restrict__ out) {
  const int K = 1024, N = 1024;
  int wave = threadIdx.x >> 5, lane = threadIdx.x & 31;
  int mblk = blockIdx.y * 64;
  int m0 = mblk + wave * 16;
  int n0 = blockIdx.x * 64;
  __shared__ __align__(16) bf16_t lA[2][64 * 32];
  __shared__ __align__(16) bf16_t lB[2][32 * 64];
  bool issuer = (threadIdx.x < 32);

  if (issuer) {
    tdm_load_2d(A + (size_t)mblk * K, &lA[0][0], 32, 64, K, K, 4096);
    tdm_load_2d(Wt + n0, &lB[0][0], 64, 32, N, N, K);
    __builtin_amdgcn_s_wait_tensorcnt(0);
  }
  __syncthreads();

  v8f acc[4];
#pragma unroll
  for (int t = 0; t < 4; ++t) acc[t] = v8f_zero();

  int cur = 0;
  for (int kk = 0; kk < K; kk += 32) {
    int nxt = cur ^ 1;
    if (issuer && (kk + 32 < K)) {
      tdm_load_2d(A + (size_t)mblk * K + (kk + 32), &lA[nxt][0], 32, 64, K, K, 4096);
      tdm_load_2d(Wt + (size_t)(kk + 32) * N + n0, &lB[nxt][0], 64, 32, N, N, K);
    }
    v16bf a = load_a_frag(&lA[cur][wave * 16 * 32], 32);
    v16bf b[4];
#pragma unroll
    for (int t = 0; t < 4; ++t) b[t] = load_b_frag_rowmajor(&lB[cur][16 * t], 64);
#pragma unroll
    for (int t = 0; t < 4; ++t) acc[t] = wmma_bf16(a, b[t], acc[t]);
    if (issuer) __builtin_amdgcn_s_wait_tensorcnt(0);
    __syncthreads();
    cur = nxt;
  }
  int hh = lane >> 4, nn = lane & 15;
#pragma unroll
  for (int t = 0; t < 4; ++t) {
    int col = n0 + 16 * t + nn;
    float bv = bias[col];
#pragma unroll
    for (int r = 0; r < 8; ++r)
      out[(size_t)(m0 + r + 8 * hh) * N + col] = acc[t][r] + bv;
  }
}

// ---------------------------------------------------------------------------
// Kernel 4: flash attention. QKV bf16 [4096,3072] (q|k|v), O bf16 [4096,1024].
// grid.x = n/64 (4 waves x 16 query rows), grid.y = b*h (32).
// ---------------------------------------------------------------------------
__global__ void msa_attn(const bf16_t* __restrict__ QKV, bf16_t* __restrict__ O) {
  const int LD = 3072, NSEQ = 2048, DH = 64;
  int wave = threadIdx.x >> 5, lane = threadIdx.x & 31;
  int bh = blockIdx.y;
  int bb = bh >> 4, h = bh & 15;
  int q0 = blockIdx.x * 64 + wave * 16;
  size_t rowbase = (size_t)bb * NSEQ;
  const bf16_t* qptr = QKV + (rowbase + q0) * LD + h * DH;
  const bf16_t* kptr = QKV + rowbase * LD + 1024 + h * DH;
  const bf16_t* vptr = QKV + rowbase * LD + 2048 + h * DH;

  v16bf qf0 = load_a_frag(qptr, LD);        // K dims 0..31
  v16bf qf1 = load_a_frag(qptr + 32, LD);   // K dims 32..63

  float mstat[8], lstat[8];
#pragma unroll
  for (int r = 0; r < 8; ++r) { mstat[r] = -1e30f; lstat[r] = 0.0f; }
  v8f o[4];
#pragma unroll
  for (int u = 0; u < 4; ++u) o[u] = v8f_zero();

  __shared__ __align__(16) bf16_t pst[4][16 * 40];
  bf16_t* myp = pst[wave];
  int hh = lane >> 4, nn = lane & 15;
  const float scale = 1.0f / 64.0f;

  for (int key0 = 0; key0 < NSEQ; key0 += 32) {
    // load all four K^T fragments first, then issue the 4 WMMAs densely
    v16bf b00 = load_b_frag_gather(kptr + (size_t)key0 * LD, LD);
    v16bf b01 = load_b_frag_gather(kptr + (size_t)key0 * LD + 32, LD);
    v16bf b10 = load_b_frag_gather(kptr + (size_t)(key0 + 16) * LD, LD);
    v16bf b11 = load_b_frag_gather(kptr + (size_t)(key0 + 16) * LD + 32, LD);
    v8f s0 = v8f_zero(), s1 = v8f_zero();
    s0 = wmma_bf16(qf0, b00, s0);
    s1 = wmma_bf16(qf0, b10, s1);
    s0 = wmma_bf16(qf1, b01, s0);
    s1 = wmma_bf16(qf1, b11, s1);

    float alpha[8];
#pragma unroll
    for (int r = 0; r < 8; ++r) {
      float a0 = s0[r] * scale, a1 = s1[r] * scale;
      float mx = fmaxf(a0, a1);
#pragma unroll
      for (int msk = 1; msk < 16; msk <<= 1)
        mx = fmaxf(mx, __shfl_xor(mx, msk, 32));
      float nm = fmaxf(mstat[r], mx);
      alpha[r] = __expf(mstat[r] - nm);
      float p0 = __expf(a0 - nm), p1 = __expf(a1 - nm);
      s0[r] = p0; s1[r] = p1;
      float rs = p0 + p1;
#pragma unroll
      for (int msk = 1; msk < 16; msk <<= 1)
        rs += __shfl_xor(rs, msk, 32);
      lstat[r] = lstat[r] * alpha[r] + rs;
      mstat[r] = nm;
    }
#pragma unroll
    for (int u = 0; u < 4; ++u)
#pragma unroll
      for (int r = 0; r < 8; ++r) o[u][r] *= alpha[r];

#pragma unroll
    for (int r = 0; r < 8; ++r) {
      myp[(r + 8 * hh) * 40 + nn]      = (bf16_t)s0[r];
      myp[(r + 8 * hh) * 40 + 16 + nn] = (bf16_t)s1[r];
    }
    __syncthreads();
    v16bf pf = load_a_frag(myp, 40);
    v16bf vf[4];
#pragma unroll
    for (int u = 0; u < 4; ++u)
      vf[u] = load_b_frag_rowmajor(vptr + (size_t)key0 * LD + u * 16, LD);
#pragma unroll
    for (int u = 0; u < 4; ++u) o[u] = wmma_bf16(pf, vf[u], o[u]);
    __syncthreads();
  }

  bf16_t* obase = O + (rowbase + q0) * 1024 + h * DH;
#pragma unroll
  for (int r = 0; r < 8; ++r) {
    float inv = 1.0f / lstat[r];
#pragma unroll
    for (int u = 0; u < 4; ++u)
      obase[(size_t)(r + 8 * hh) * 1024 + u * 16 + nn] = (bf16_t)(o[u][r] * inv);
  }
}

// ---------------------------------------------------------------------------
extern "C" void kernel_launch(void* const* d_in, const int* in_sizes, int n_in,
                              void* d_out, int out_size, void* d_ws, size_t ws_size,
                              hipStream_t stream) {
  const float* x     = (const float*)d_in[0];  // [2,2048,1024]
  const float* w_qkv = (const float*)d_in[1];  // [3072,1024]
  const float* w_out = (const float*)d_in[2];  // [1024,1024]
  const float* b_out = (const float*)d_in[3];  // [1024]
  float* out = (float*)d_out;                  // [2,2048,1024] f32

  const int BN = 4096, DIM = 1024, TD = 3072;
  char* w = (char*)d_ws;
  bf16_t* xb    = (bf16_t*)(w);                      //  8 MB  X bf16
  bf16_t* wqkvt = (bf16_t*)(w + ((size_t)8  << 20)); //  6 MB  Wqkv^T bf16 [1024,3072]
  bf16_t* woutt = (bf16_t*)(w + ((size_t)14 << 20)); //  2 MB  Wout^T bf16 [1024,1024]
  bf16_t* qkv   = (bf16_t*)(w + ((size_t)16 << 20)); // 24 MB  QKV bf16 [4096,3072]
  bf16_t* obuf  = (bf16_t*)(w + ((size_t)40 << 20)); //  8 MB  attn out bf16 [4096,1024]

  msa_cvt_bf16<<<(BN * DIM) / 256, 256, 0, stream>>>(x, xb, BN * DIM);
  msa_transpose_cvt<<<dim3(DIM / 32, TD / 32), dim3(32, 8), 0, stream>>>(w_qkv, wqkvt, TD, DIM);
  msa_transpose_cvt<<<dim3(DIM / 32, DIM / 32), dim3(32, 8), 0, stream>>>(w_out, woutt, DIM, DIM);
  msa_qkv_gemm<<<dim3(TD / 64, BN / 64), 128, 0, stream>>>(xb, wqkvt, qkv);
  msa_attn<<<dim3(2048 / 64, 32), 128, 0, stream>>>(qkv, obuf);
  msa_out_gemm<<<dim3(DIM / 64, BN / 64), 128, 0, stream>>>(obuf, woutt, b_out, out);
}
